// AutoregressiveHDFormer_23502061043851
// MI455X (gfx1250) — 16-bit and 8-bit matrix operand order repaired and checked by reading, compiled, not run on hardware
//
#include <hip/hip_runtime.h>
#include <hip/hip_bf16.h>


#define BB     4
#define LSEQ   1024
#define DMOD   512
#define NVOCAB 32000

typedef __attribute__((ext_vector_type(16))) __bf16 v16bf;
typedef __attribute__((ext_vector_type(8)))  float  v8f;
typedef __attribute__((ext_vector_type(8)))  int    v8i;

__device__ __forceinline__ unsigned short f2bf(float f) {
  unsigned int u = __builtin_bit_cast(unsigned int, f);
  u += 0x7FFFu + ((u >> 16) & 1u);
  return (unsigned short)(u >> 16);
}

__device__ __forceinline__ unsigned lds_off(const void* p) {
  return (unsigned)(unsigned long long)p;
}

__global__ __launch_bounds__(256) void embed_kernel(
    const int* __restrict__ tok, const float* __restrict__ table,
    float* __restrict__ out) {
  long idx = (long)blockIdx.x * 256 + threadIdx.x;
  int d = (int)(idx & (DMOD - 1));
  long r = idx >> 9;
  int l = (int)(r & (LSEQ - 1));
  int tk = tok[r];
  int de = d & ~1;
  float rate = __expf(-(float)de * (9.210340371976184f / (float)DMOD));
  float ang = (float)l * rate;
  float pe = (d & 1) ? __cosf(ang) : __sinf(ang);
  out[idx] = table[(long)tk * DMOD + d] * 22.62741699796952f + pe;
}

__global__ __launch_bounds__(256) void ln_kernel(
    const float* __restrict__ x, const float* __restrict__ res,
    const float* __restrict__ g, const float* __restrict__ beta,
    float* __restrict__ y) {
  __shared__ float rs[256], rq[256];
  const long row = blockIdx.x;
  const int t = threadIdx.x;
  const float* xr = x + row * DMOD;
  float v0 = xr[t], v1 = xr[t + 256];
  if (res) { v0 += res[row * DMOD + t]; v1 += res[row * DMOD + t + 256]; }
  rs[t] = v0 + v1;
  rq[t] = v0 * v0 + v1 * v1;
  __syncthreads();
  for (int o = 128; o > 0; o >>= 1) {
    if (t < o) { rs[t] += rs[t + o]; rq[t] += rq[t + o]; }
    __syncthreads();
  }
  float mean = rs[0] * (1.0f / DMOD);
  float var  = rq[0] * (1.0f / DMOD) - mean * mean;
  float rstd = rsqrtf(var + 1e-6f);
  y[row * DMOD + t]       = (v0 - mean) * rstd * g[t]       + beta[t];
  y[row * DMOD + t + 256] = (v1 - mean) * rstd * g[t + 256] + beta[t + 256];
}

__global__ __launch_bounds__(256) void sign_kernel(
    const float* __restrict__ x, signed char* __restrict__ s,
    int* __restrict__ cnt) {
  __shared__ int red[256];
  const long row = blockIdx.x;
  const int t = threadIdx.x;
  int c = 0;
  for (int e = 0; e < 2; ++e) {
    long i = row * DMOD + t + e * 256;
    float f = x[i];
    signed char sv = (f > 0.f) ? 1 : ((f < 0.f) ? -1 : 0);
    s[i] = sv;
    c += (sv != 0);
  }
  if (cnt) {
    red[t] = c;
    __syncthreads();
    for (int o = 128; o > 0; o >>= 1) {
      if (t < o) red[t] += red[t + o];
      __syncthreads();
    }
    if (t == 0) cnt[row] = red[0];
  }
}

__global__ __launch_bounds__(256) void softmax_mask_kernel(
    float* __restrict__ attn, const int* __restrict__ tok, int mode) {
  __shared__ float red[256];
  const int i = blockIdx.x, b = blockIdx.y, t = threadIdx.x;
  float* row = attn + ((long)b * LSEQ + i) * LSEQ;
  float v[4], mx = -1e30f;
  for (int e = 0; e < 4; ++e) { v[e] = row[t + e * 256]; mx = fmaxf(mx, v[e]); }
  red[t] = mx; __syncthreads();
  for (int o = 128; o > 0; o >>= 1) { if (t < o) red[t] = fmaxf(red[t], red[t + o]); __syncthreads(); }
  mx = red[0]; __syncthreads();
  float sum = 0.f;
  for (int e = 0; e < 4; ++e) { v[e] = __expf(v[e] - mx); sum += v[e]; }
  red[t] = sum; __syncthreads();
  for (int o = 128; o > 0; o >>= 1) { if (t < o) red[t] += red[t + o]; __syncthreads(); }
  float inv = 1.0f / red[0];
  for (int e = 0; e < 4; ++e) {
    int j = t + e * 256;
    float m = (mode == 0) ? ((tok[(long)b * LSEQ + j] == 0) ? 1.f : 0.f)
                          : ((j > i) ? 1.f : 0.f);
    row[j] = v[e] * inv * m;
  }
}

enum { EPI_BIAS = 0, EPI_BIAS_RELU = 1, EPI_TANH_ANCHOR = 2, EPI_BIAS_NT = 3 };

template <bool TRANS_A, int EPI>
__global__ __launch_bounds__(256) void gemm_bf16(
    const float* __restrict__ A, const float* __restrict__ Bm,
    float* __restrict__ C, int K, int lda, int ldb, int ldc,
    long sA, long sB, long sC,
    const float* __restrict__ bias, const float* __restrict__ aux, int auxld) {
  constexpr int BM = 128, BN = 64, BK = 32, AST = 34, BST = 34;
  __shared__ __align__(16) unsigned short At[BM * AST];
  __shared__ __align__(16) unsigned short Bt[BN * BST];
  const int bz = blockIdx.z;
  A += (long)bz * sA; Bm += (long)bz * sB; C += (long)bz * sC;
  const int row0 = blockIdx.y * BM, col0 = blockIdx.x * BN;
  const int t = threadIdx.x, lane = t & 31, wave = t >> 5;
  const int wm = (wave >> 1) * 32, wn = (wave & 1) * 32;
  const int hl = lane & 15, hh = lane >> 4;
  v8f acc[2][2] = {};
  for (int k0 = 0; k0 < K; k0 += BK) {
    __syncthreads();
    if (!TRANS_A) {
      for (int e = 0; e < 4; ++e) {
        int f = e * 256 + t;
        int r = f >> 3, c4 = (f & 7) << 2;
        const float4 x = *(const float4*)&A[(long)(row0 + r) * lda + (k0 + c4)];
        unsigned u0 = (unsigned)f2bf(x.x) | ((unsigned)f2bf(x.y) << 16);
        unsigned u1 = (unsigned)f2bf(x.z) | ((unsigned)f2bf(x.w) << 16);
        *(unsigned*)&At[r * AST + c4]     = u0;
        *(unsigned*)&At[r * AST + c4 + 2] = u1;
      }
    } else {
      for (int e = 0; e < 4; ++e) {
        int f = e * 256 + t;
        int k = f >> 5, m4 = (f & 31) << 2;
        const float4 x = *(const float4*)&A[(long)(k0 + k) * lda + (row0 + m4)];
        At[(m4 + 0) * AST + k] = f2bf(x.x);
        At[(m4 + 1) * AST + k] = f2bf(x.y);
        At[(m4 + 2) * AST + k] = f2bf(x.z);
        At[(m4 + 3) * AST + k] = f2bf(x.w);
      }
    }
    for (int e = 0; e < 2; ++e) {
      int f = e * 256 + t;
      int k = f >> 4, n4 = (f & 15) << 2;
      const float4 x = *(const float4*)&Bm[(long)(k0 + k) * ldb + (col0 + n4)];
      Bt[(n4 + 0) * BST + k] = f2bf(x.x);
      Bt[(n4 + 1) * BST + k] = f2bf(x.y);
      Bt[(n4 + 2) * BST + k] = f2bf(x.z);
      Bt[(n4 + 3) * BST + k] = f2bf(x.w);
    }
    __syncthreads();
    if (k0 + BK < K) {
      const float* pa = TRANS_A ? &A[(long)(k0 + BK) * lda + row0]
                                : &A[(long)row0 * lda + (k0 + BK)];
      __builtin_prefetch(pa + t * 16, 0, 1);
      __builtin_prefetch(&Bm[(long)(k0 + BK) * ldb + col0] + t * 8, 0, 1);
    }
    union Frag { unsigned int u[8]; v16bf v; } af[2], bfr[2];
    for (int tm = 0; tm < 2; ++tm) {
      int m = wm + tm * 16 + hl;
      for (int vi = 0; vi < 8; ++vi) {
        int kk = ((vi & 4) << 2) + (hh << 3) + ((vi & 3) << 1);
        af[tm].u[vi] = *(const unsigned int*)&At[m * AST + kk];
      }
    }
    for (int tn = 0; tn < 2; ++tn) {
      int n = wn + tn * 16 + hl;
      for (int vi = 0; vi < 8; ++vi) {
        int kk = ((vi & 4) << 2) + (hh << 3) + ((vi & 3) << 1);
        bfr[tn].u[vi] = *(const unsigned int*)&Bt[n * BST + kk];
      }
    }
    for (int tm = 0; tm < 2; ++tm)
      for (int tn = 0; tn < 2; ++tn)
        acc[tm][tn] = __builtin_amdgcn_wmma_f32_16x16x32_bf16(
            false, af[tm].v, false, bfr[tn].v, (short)0, acc[tm][tn], false, false);
  }
  for (int tm = 0; tm < 2; ++tm)
    for (int tn = 0; tn < 2; ++tn) {
      int colg = col0 + wn + tn * 16 + hl;
      for (int r = 0; r < 8; ++r) {
        int rowg = row0 + wm + tm * 16 + hh * 8 + r;
        float v = acc[tm][tn][r];
        if (EPI == EPI_BIAS || EPI == EPI_BIAS_NT) v += bias[colg];
        else if (EPI == EPI_BIAS_RELU) { v += bias[colg]; v = v > 0.f ? v : 0.f; }
        else v = tanhf(v * aux[(long)rowg * auxld + colg]);
        if (EPI == EPI_BIAS_NT)
          __builtin_nontemporal_store(v, &C[(long)rowg * ldc + colg]);
        else
          C[(long)rowg * ldc + colg] = v;
      }
    }
}

__global__ __launch_bounds__(256) void gemm_sign_logits(
    const signed char* __restrict__ SK, const signed char* __restrict__ SQ,
    const int* __restrict__ sumsq, float* __restrict__ Out) {
  constexpr int BM = 128, BN = 64, BK = 64, AST = 80, BST = 80;
  constexpr int NK = DMOD / BK;
  __shared__ __align__(16) signed char At[2][BM * AST];
  __shared__ __align__(16) signed char Bt[2][BN * BST];
  const int bz = blockIdx.z;
  SK += (long)bz * LSEQ * DMOD;
  SQ += (long)bz * LSEQ * DMOD;
  sumsq += (long)bz * LSEQ;
  Out += (long)bz * LSEQ * LSEQ;
  const int row0 = blockIdx.y * BM, col0 = blockIdx.x * BN;
  const int t = threadIdx.x, lane = t & 31, wave = t >> 5;
  const int wm = (wave >> 1) * 32, wn = (wave & 1) * 32;
  const int hl = lane & 15, hh = lane >> 4;

  const int rA = t >> 2, c16 = (t & 3) << 4;
  const signed char* gA = SK + (long)(row0 + rA) * DMOD + c16;
  const signed char* gB = SQ + (long)(col0 + rA) * DMOD + c16;
  unsigned lA0[2], lA1[2], lB0[2];
  for (int p = 0; p < 2; ++p) {
    lA0[p] = lds_off(&At[p][rA * AST + c16]);
    lA1[p] = lds_off(&At[p][(rA + 64) * AST + c16]);
    lB0[p] = lds_off(&Bt[p][rA * BST + c16]);
  }
  auto issue = [&](int p, int kbyte) {
    const signed char* a0 = gA + kbyte;
    const signed char* a1 = gA + 64 * DMOD + kbyte;
    const signed char* b0 = gB + kbyte;
    asm volatile("global_load_async_to_lds_b128 %0, %1, off"
                 :: "v"(lA0[p]), "v"(a0) : "memory");
    asm volatile("global_load_async_to_lds_b128 %0, %1, off"
                 :: "v"(lA1[p]), "v"(a1) : "memory");
    asm volatile("global_load_async_to_lds_b128 %0, %1, off"
                 :: "v"(lB0[p]), "v"(b0) : "memory");
  };

  issue(0, 0);
  v8i acc[2][2] = {};
#pragma unroll
  for (int i = 0; i < NK; ++i) {
    const int cur = i & 1;
    if (i + 1 < NK) {
      issue(cur ^ 1, (i + 1) * BK);
      asm volatile("s_wait_asynccnt 3" ::: "memory");
    } else {
      asm volatile("s_wait_asynccnt 0" ::: "memory");
    }
    __syncthreads();
    union Frag { int u[8]; v8i v; } af[2], bfr[2];
    for (int tm = 0; tm < 2; ++tm) {
      int m = wm + tm * 16 + hl;
      for (int vi = 0; vi < 8; ++vi) {
        int kk = (((vi >> 1) & 1) << 4) + (((vi >> 2) & 1) << 5) + ((vi & 1) << 2) + (hh << 3);
        af[tm].u[vi] = *(const int*)&At[cur][m * AST + kk];
      }
    }
    for (int tn = 0; tn < 2; ++tn) {
      int n = wn + tn * 16 + hl;
      for (int vi = 0; vi < 8; ++vi) {
        int kk = (((vi >> 1) & 1) << 4) + (((vi >> 2) & 1) << 5) + ((vi & 1) << 2) + (hh << 3);
        bfr[tn].u[vi] = *(const int*)&Bt[cur][n * BST + kk];
      }
    }
    for (int tm = 0; tm < 2; ++tm)
      for (int tn = 0; tn < 2; ++tn)
        acc[tm][tn] = __builtin_amdgcn_wmma_i32_16x16x64_iu8(
            true, af[tm].v, true, bfr[tn].v, acc[tm][tn], false, false);
    __syncthreads();
  }
  for (int tm = 0; tm < 2; ++tm)
    for (int tn = 0; tn < 2; ++tn) {
      int colg = col0 + wn + tn * 16 + hl;
      int add = sumsq[colg];
      for (int r = 0; r < 8; ++r) {
        int rowg = row0 + wm + tm * 16 + hh * 8 + r;
        Out[(long)rowg * LSEQ + colg] = (float)(acc[tm][tn][r] + add) * (1.0f / 1024.0f);
      }
    }
}

struct MhaW { const float *wq,*bq,*wk,*bk,*wv,*bv,*wo,*bo,*anchor; };

static void run_mha(hipStream_t stream,
                    const float* q_in, const float* k_in, const float* v_in,
                    int maskmode, const int* tok, const MhaW& w, float* out,
                    float* tq, float* tk, float* tv, float* tctx,
                    float* attnmat, signed char* sq8, signed char* sk8, int* sumsq) {
  const int Mall = BB * LSEQ;
  dim3 gP(DMOD / 64, Mall / 128, 1);
  gemm_bf16<false, EPI_BIAS><<<gP, 256, 0, stream>>>(
      q_in, w.wq, tq, DMOD, DMOD, DMOD, DMOD, 0, 0, 0, w.bq, nullptr, 0);
  gemm_bf16<false, EPI_BIAS><<<gP, 256, 0, stream>>>(
      k_in, w.wk, tk, DMOD, DMOD, DMOD, DMOD, 0, 0, 0, w.bk, nullptr, 0);
  gemm_bf16<false, EPI_BIAS><<<gP, 256, 0, stream>>>(
      v_in, w.wv, tv, DMOD, DMOD, DMOD, DMOD, 0, 0, 0, w.bv, nullptr, 0);
  sign_kernel<<<Mall, 256, 0, stream>>>(tq, sq8, sumsq);
  sign_kernel<<<Mall, 256, 0, stream>>>(tk, sk8, nullptr);
  gemm_sign_logits<<<dim3(LSEQ / 64, LSEQ / 128, BB), 256, 0, stream>>>(
      sk8, sq8, sumsq, attnmat);
  softmax_mask_kernel<<<dim3(LSEQ, BB), 256, 0, stream>>>(attnmat, tok, maskmode);
  gemm_bf16<true, EPI_TANH_ANCHOR><<<dim3(DMOD / 64, LSEQ / 128, BB), 256, 0, stream>>>(
      attnmat, tv, tctx, LSEQ, LSEQ, DMOD, DMOD,
      (long)LSEQ * LSEQ, (long)LSEQ * DMOD, (long)LSEQ * DMOD,
      nullptr, w.anchor, DMOD);
  gemm_bf16<false, EPI_BIAS><<<gP, 256, 0, stream>>>(
      tctx, w.wo, out, DMOD, DMOD, DMOD, DMOD, 0, 0, 0, w.bo, nullptr, 0);
}

extern "C" void kernel_launch(void* const* d_in, const int* in_sizes, int n_in,
                              void* d_out, int out_size, void* d_ws, size_t ws_size,
                              hipStream_t stream) {
  (void)in_sizes; (void)n_in; (void)out_size; (void)ws_size;
  int ix = 0;
  auto F = [&]() -> const float* { return (const float*)d_in[ix++]; };
  const int* inp = (const int*)d_in[ix++];
  const int* tar = (const int*)d_in[ix++];
  const float* emb_enc = F();
  const float* emb_dec = F();
  struct EncL { MhaW mha; const float *w1,*b1,*w2,*b2,*ln1g,*ln1b,*ln2g,*ln2b; } enc[2];
  for (int l = 0; l < 2; ++l) {
    enc[l].mha = MhaW{F(), F(), F(), F(), F(), F(), F(), F(), F()};
    enc[l].w1 = F(); enc[l].b1 = F(); enc[l].w2 = F(); enc[l].b2 = F();
    enc[l].ln1g = F(); enc[l].ln1b = F(); enc[l].ln2g = F(); enc[l].ln2b = F();
  }
  struct DecL { MhaW m1, m2;
                const float *w1,*b1,*w2,*b2,*ln1g,*ln1b,*ln2g,*ln2b,*ln3g,*ln3b; } dec[2];
  for (int l = 0; l < 2; ++l) {
    dec[l].m1 = MhaW{F(), F(), F(), F(), F(), F(), F(), F(), F()};
    dec[l].m2 = MhaW{F(), F(), F(), F(), F(), F(), F(), F(), F()};
    dec[l].w1 = F(); dec[l].b1 = F(); dec[l].w2 = F(); dec[l].b2 = F();
    dec[l].ln1g = F(); dec[l].ln1b = F(); dec[l].ln2g = F(); dec[l].ln2b = F();
    dec[l].ln3g = F(); dec[l].ln3b = F();
  }
  const float* fw = F();
  const float* fb = F();

  const size_t SZ = (size_t)BB * LSEQ * DMOD * sizeof(float);
  char* w = (char*)d_ws;
  float* xbuf = (float*)(w + 0 * SZ);
  float* ybuf = (float*)(w + 1 * SZ);
  float* t0   = (float*)(w + 2 * SZ);
  float* t1   = (float*)(w + 3 * SZ);
  float* t2   = (float*)(w + 4 * SZ);
  float* t3   = (float*)(w + 5 * SZ);
  float* t4   = (float*)(w + 6 * SZ);
  float* t5   = (float*)(w + 7 * SZ);
  float* attnmat = (float*)(w + 8 * SZ);
  signed char* sq8 = (signed char*)(w + 10 * SZ);
  signed char* sk8 = sq8 + (size_t)BB * LSEQ * DMOD;
  int* sumsq = (int*)(w + 10 * SZ + 2 * (size_t)BB * LSEQ * DMOD);

  const int Mall = BB * LSEQ;
  dim3 gFFN(DMOD / 64, Mall / 128, 1);

  embed_kernel<<<(Mall * DMOD) / 256, 256, 0, stream>>>(inp, emb_enc, xbuf);
  for (int l = 0; l < 2; ++l) {
    run_mha(stream, xbuf, xbuf, xbuf,  0, inp, enc[l].mha, t4,
            t0, t1, t2, t3, attnmat, sq8, sk8, sumsq);
    ln_kernel<<<Mall, 256, 0, stream>>>(t4, xbuf, enc[l].ln1g, enc[l].ln1b, t5);
    gemm_bf16<false, EPI_BIAS_RELU><<<gFFN, 256, 0, stream>>>(
        t5, enc[l].w1, t0, DMOD, DMOD, DMOD, DMOD, 0, 0, 0, enc[l].b1, nullptr, 0);
    gemm_bf16<false, EPI_BIAS><<<gFFN, 256, 0, stream>>>(
        t0, enc[l].w2, t1, DMOD, DMOD, DMOD, DMOD, 0, 0, 0, enc[l].b2, nullptr, 0);
    ln_kernel<<<Mall, 256, 0, stream>>>(t1, nullptr, enc[l].ln2g, enc[l].ln2b, xbuf);
  }

  embed_kernel<<<(Mall * DMOD) / 256, 256, 0, stream>>>(tar, emb_dec, ybuf);
  for (int l = 0; l < 2; ++l) {
    run_mha(stream, ybuf, ybuf, ybuf,  1, nullptr, dec[l].m1, t4,
            t0, t1, t2, t3, attnmat, sq8, sk8, sumsq);
    ln_kernel<<<Mall, 256, 0, stream>>>(t4, ybuf, dec[l].ln1g, dec[l].ln1b, t5);
    run_mha(stream,  t5,  xbuf,  xbuf,  1, nullptr,
            dec[l].m2, t4, t0, t1, t2, t3, attnmat, sq8, sk8, sumsq);
    ln_kernel<<<Mall, 256, 0, stream>>>(t4, t5, dec[l].ln2g, dec[l].ln2b, t0);
    gemm_bf16<false, EPI_BIAS_RELU><<<gFFN, 256, 0, stream>>>(
        t0, dec[l].w1, t1, DMOD, DMOD, DMOD, DMOD, 0, 0, 0, dec[l].b1, nullptr, 0);
    gemm_bf16<false, EPI_BIAS><<<gFFN, 256, 0, stream>>>(
        t1, dec[l].w2, t2, DMOD, DMOD, DMOD, DMOD, 0, 0, 0, dec[l].b2, nullptr, 0);
    ln_kernel<<<Mall, 256, 0, stream>>>(t2, nullptr, dec[l].ln3g, dec[l].ln3b, ybuf);
  }

  gemm_bf16<false, EPI_BIAS_NT><<<dim3(NVOCAB / 64, Mall / 128, 1), 256, 0, stream>>>(
      ybuf, fw, (float*)d_out, DMOD, DMOD, NVOCAB, NVOCAB, 0, 0, 0, fb, nullptr, 0);
}
